// Phi3DecoderLayer_22316650070863
// MI455X (gfx1250) — matrix operand order repaired and checked by reading, not compiled, not run on hardware
//
#include <hip/hip_runtime.h>
#include <hip/hip_bf16.h>

#define B_  4
#define S_  1024
#define H_  3072
#define NH_ 32
#define D_  96
#define I_  8192

typedef __bf16 bf16;
typedef __attribute__((ext_vector_type(16))) __bf16 v16bf;
typedef __attribute__((ext_vector_type(8)))  __bf16 v8bf;
typedef __attribute__((ext_vector_type(8)))  float  v8f;

__device__ inline bf16 f2bf(float f) {
    union { float f; unsigned u; } a; a.f = f;
    unsigned r = a.u + 0x7fffu + ((a.u >> 16) & 1u);
    union { unsigned short s; bf16 b; } o; o.s = (unsigned short)(r >> 16);
    return o.b;
}

__device__ inline v16bf mk16(v8bf lo, v8bf hi) {
    v16bf r;
#pragma unroll
    for (int i = 0; i < 8; ++i) { r[i] = lo[i]; r[i + 8] = hi[i]; }
    return r;
}

__device__ inline float gelu_tanh(float x) {
    const float c = 0.7978845608028654f;
    return 0.5f * x * (1.0f + tanhf(c * (x + 0.044715f * x * x * x)));
}

// ---------------- fp32 -> bf16 weight conversion ----------------
__global__ __launch_bounds__(256) void cvt_kernel(const float* __restrict__ src,
                                                  bf16* __restrict__ dst, int n) {
    for (int i = blockIdx.x * blockDim.x + threadIdx.x; i < n;
         i += gridDim.x * blockDim.x)
        dst[i] = f2bf(src[i]);
}

// ---------------- LayerNorm: fp32 in -> bf16 out ----------------
__global__ __launch_bounds__(256) void ln_kernel(const float* __restrict__ x,
                                                 const float* __restrict__ gw,
                                                 const float* __restrict__ bw,
                                                 bf16* __restrict__ out) {
    __shared__ float red0[256];
    __shared__ float red1[256];
    const int row = blockIdx.x;
    const int tid = threadIdx.x;
    const float* xr = x + (size_t)row * H_;
    float s = 0.f, ss = 0.f;
    for (int i = tid; i < H_; i += 256) { float v = xr[i]; s += v; ss += v * v; }
    red0[tid] = s; red1[tid] = ss;
    __syncthreads();
    for (int st = 128; st > 0; st >>= 1) {
        if (tid < st) { red0[tid] += red0[tid + st]; red1[tid] += red1[tid + st]; }
        __syncthreads();
    }
    const float mean = red0[0] * (1.0f / H_);
    const float var  = red1[0] * (1.0f / H_) - mean * mean;
    const float inv  = rsqrtf(var + 1e-5f);
    bf16* orow = out + (size_t)row * H_;
    for (int i = tid; i < H_; i += 256)
        orow[i] = f2bf((xr[i] - mean) * inv * gw[i] + bw[i]);
}

// ---------------- bf16 WMMA GEMM with fused epilogue ----------------
// C[M,N] = A[M,K] * B[K,N] (+bias) (gelu) (+res) -> f32 or bf16
// A tile is staged into LDS with gfx1250 async global->LDS DMA (ASYNCcnt).
template <bool BIAS, bool GELU, bool RES, bool OUTBF>
__global__ __launch_bounds__(256) void gemm_bf16_kernel(
    const bf16* __restrict__ A, const bf16* __restrict__ Bw,
    const float* __restrict__ bias, const float* __restrict__ res,
    float* __restrict__ outf, bf16* __restrict__ outb, int M, int N, int K) {
    constexpr int BM = 128, BN = 128, BK = 32;
    constexpr int AST = BK + 8;  // 40 elems, 80B row stride (16B aligned)
    constexpr int BST = BK + 8;
    __shared__ __align__(16) bf16 As[BM * AST];
    __shared__ __align__(16) bf16 Bs[BN * BST];

    const int tid  = threadIdx.x;
    const int lane = tid & 31;
    const int wave = tid >> 5;
    const int g    = lane >> 4;
    const int lm   = lane & 15;
    const int waveM = (wave >> 2) * 64;  // 0 or 64
    const int waveN = (wave & 3) * 32;   // 0,32,64,96
    const int bm = blockIdx.y * BM, bn = blockIdx.x * BN;

    const int arow = tid >> 1, akh = tid & 1;  // A stage: row, 16-elem half
    const int bk = tid & 31, bng = tid >> 5;   // B stage: k row, 16-col group

    // per-thread LDS byte address for the A-tile async DMA destination
    const unsigned aLds = (unsigned)(size_t)&As[arow * AST + akh * 16];

    v8f acc[4][2] = {};

    for (int k0 = 0; k0 < K; k0 += BK) {
        // ---- stage A tile via async global->LDS DMA (32B per thread) ----
        {
            const unsigned aOff =
                (unsigned)(((size_t)(bm + arow) * K + (size_t)k0 + akh * 16) * 2);
            asm volatile(
                "global_load_async_to_lds_b128 %0, %1, %2\n\t"
                "global_load_async_to_lds_b128 %0, %1, %2 offset:16"
                :
                : "v"(aLds), "v"(aOff), "s"(A)
                : "memory");
        }
        // ---- stage B tile transposed: Bs[n][k] ----
        {
            const bf16* bp = Bw + (size_t)(k0 + bk) * N + bn + bng * 16;
            v8bf b0 = *(const v8bf*)bp;
            v8bf b1 = *(const v8bf*)(bp + 8);
#pragma unroll
            for (int i = 0; i < 8; ++i) {
                Bs[(bng * 16 + i) * BST + bk]     = b0[i];
                Bs[(bng * 16 + 8 + i) * BST + bk] = b1[i];
            }
        }
        if (k0 + BK < K) {  // gfx1250 global_prefetch path for next B tile
            __builtin_prefetch(Bw + (size_t)(k0 + BK + bk) * N + bn + bng * 16, 0, 0);
        }
        asm volatile("s_wait_asynccnt 0" ::: "memory");
        __syncthreads();

        v16bf af[4], bfr[2];
#pragma unroll
        for (int mi = 0; mi < 4; ++mi) {  // A layout: lane=row, K interleaved at 8
            const bf16* p = &As[(waveM + mi * 16 + lm) * AST];
            af[mi] = mk16(*(const v8bf*)(p + g * 8), *(const v8bf*)(p + 16 + g * 8));
        }
#pragma unroll
        for (int ni = 0; ni < 2; ++ni) {  // B layout: lane=col, K interleaved at 8
            const bf16* p = &Bs[(waveN + ni * 16 + lm) * BST];
            bfr[ni] = mk16(*(const v8bf*)(p + g * 8), *(const v8bf*)(p + 16 + g * 8));
        }
#pragma unroll
        for (int mi = 0; mi < 4; ++mi)
#pragma unroll
            for (int ni = 0; ni < 2; ++ni)
                acc[mi][ni] = __builtin_amdgcn_wmma_f32_16x16x32_bf16(
                    false, af[mi], false, bfr[ni], (short)0, acc[mi][ni], false, false);
        __syncthreads();
    }

    // epilogue: C layout -> lane L: row = r + 8*(L>>4), col = L&15
#pragma unroll
    for (int mi = 0; mi < 4; ++mi)
#pragma unroll
        for (int ni = 0; ni < 2; ++ni)
#pragma unroll
            for (int r = 0; r < 8; ++r) {
                const int row = bm + waveM + mi * 16 + r + 8 * g;
                const int col = bn + waveN + ni * 16 + lm;
                float v = acc[mi][ni][r];
                if (BIAS) v += bias[col];
                if (GELU) v = gelu_tanh(v);
                const size_t o = (size_t)row * N + col;
                if (RES) v += res[o];
                if (OUTBF) outb[o] = f2bf(v);
                else       outf[o] = v;
            }
}

// ---------------- flash attention (causal), one wave per 16-row Q tile ------
__global__ __launch_bounds__(128) void attn_kernel(const bf16* __restrict__ Q,
                                                   const bf16* __restrict__ Kk,
                                                   const bf16* __restrict__ V,
                                                   bf16* __restrict__ O) {
    constexpr int VSTR = 40;  // Vt[96][32] padded
    constexpr int PSTR = 40;  // P [16][32] padded
    __shared__ __align__(16) bf16 VtS[4][D_ * VSTR];
    __shared__ __align__(16) bf16 PlS[4][16 * PSTR];

    const int lane = threadIdx.x & 31;
    const int wave = threadIdx.x >> 5;
    bf16* Vt = VtS[wave];
    bf16* Pl = PlS[wave];
    const int g = lane >> 4, lm = lane & 15;

    const int tile = blockIdx.x * 4 + wave;  // over B*NH*(S/16)
    const int QT = S_ / 16;
    const int qt = tile % QT;
    const int bh = tile / QT;
    const int head = bh % NH_;
    const int b = bh / NH_;
    const int q0 = qt * 16;
    const size_t base = (size_t)b * S_ * H_ + (size_t)head * D_;

    // Q fragments (persistent): 3 slices of K=32 over D=96
    v16bf qf[3];
    {
        const bf16* qp = Q + base + (size_t)(q0 + lm) * H_;
#pragma unroll
        for (int ds = 0; ds < 3; ++ds)
            qf[ds] = mk16(*(const v8bf*)(qp + ds * 32 + g * 8),
                          *(const v8bf*)(qp + ds * 32 + 16 + g * 8));
    }

    v8f acc[6] = {};
    float m[8], l[8];
#pragma unroll
    for (int r = 0; r < 8; ++r) { m[r] = -1e30f; l[r] = 0.f; }
    const float scale = 0.10206207261596577f;  // 1/sqrt(96)

    const int cmax = (q0 + 15) >> 5;
    for (int c = 0; c <= cmax; ++c) {
        const int k0 = c * 32;
        // stage V transposed: Vt[d][key]
        {
            const bf16* vp = V + base + (size_t)(k0 + lane) * H_;
#pragma unroll
            for (int d0 = 0; d0 < D_; d0 += 8) {
                v8bf v = *(const v8bf*)(vp + d0);
#pragma unroll
                for (int j = 0; j < 8; ++j) Vt[(d0 + j) * VSTR + lane] = v[j];
            }
        }
        // scores S = Q * K^T : two 16-key accumulators
        v8f s0 = {}, s1 = {};
        {
            const bf16* kp0 = Kk + base + (size_t)(k0 + lm) * H_;
            const bf16* kp1 = Kk + base + (size_t)(k0 + 16 + lm) * H_;
#pragma unroll
            for (int ds = 0; ds < 3; ++ds) {
                v16bf kf = mk16(*(const v8bf*)(kp0 + ds * 32 + g * 8),
                                *(const v8bf*)(kp0 + ds * 32 + 16 + g * 8));
                s0 = __builtin_amdgcn_wmma_f32_16x16x32_bf16(false, qf[ds], false, kf,
                                                             (short)0, s0, false, false);
                kf = mk16(*(const v8bf*)(kp1 + ds * 32 + g * 8),
                          *(const v8bf*)(kp1 + ds * 32 + 16 + g * 8));
                s1 = __builtin_amdgcn_wmma_f32_16x16x32_bf16(false, qf[ds], false, kf,
                                                             (short)0, s1, false, false);
            }
        }
        // online softmax per row (rows live in 16-lane half-waves)
#pragma unroll
        for (int r = 0; r < 8; ++r) {
            const int q = q0 + r + 8 * g;
            float v0 = s0[r] * scale; if (k0 + lm > q)      v0 = -1e30f;
            float v1 = s1[r] * scale; if (k0 + 16 + lm > q) v1 = -1e30f;
            float mx = fmaxf(v0, v1);
#pragma unroll
            for (int off = 1; off < 16; off <<= 1)
                mx = fmaxf(mx, __shfl_xor(mx, off, 32));
            const float mn = fmaxf(m[r], mx);
            const float p0 = __expf(v0 - mn), p1 = __expf(v1 - mn);
            float rs = p0 + p1;
#pragma unroll
            for (int off = 1; off < 16; off <<= 1) rs += __shfl_xor(rs, off, 32);
            const float corr = __expf(m[r] - mn);
            m[r] = mn;
            l[r] = l[r] * corr + rs;
#pragma unroll
            for (int nt = 0; nt < 6; ++nt) acc[nt][r] *= corr;
            Pl[(r + 8 * g) * PSTR + lm]      = f2bf(p0);
            Pl[(r + 8 * g) * PSTR + 16 + lm] = f2bf(p1);
        }
        asm volatile("s_wait_dscnt 0" ::: "memory");  // LDS RAW within wave
        // P fragment (A layout) from LDS
        const bf16* pp = &Pl[lm * PSTR];
        v16bf pf = mk16(*(const v8bf*)(pp + g * 8), *(const v8bf*)(pp + 16 + g * 8));
        // acc += P * V
#pragma unroll
        for (int nt = 0; nt < 6; ++nt) {
            const bf16* vp = &Vt[(nt * 16 + lm) * VSTR];
            v16bf vf = mk16(*(const v8bf*)(vp + g * 8), *(const v8bf*)(vp + 16 + g * 8));
            acc[nt] = __builtin_amdgcn_wmma_f32_16x16x32_bf16(false, pf, false, vf,
                                                              (short)0, acc[nt], false, false);
        }
    }
    // normalize + write ctx (bf16)
#pragma unroll
    for (int nt = 0; nt < 6; ++nt)
#pragma unroll
        for (int r = 0; r < 8; ++r) {
            const float o = acc[nt][r] / l[r];
            O[base + (size_t)(q0 + r + 8 * g) * H_ + nt * 16 + lm] = f2bf(o);
        }
}

// ---------------- host orchestration ----------------
extern "C" void kernel_launch(void* const* d_in, const int* in_sizes, int n_in,
                              void* d_out, int out_size, void* d_ws, size_t ws_size,
                              hipStream_t stream) {
    (void)in_sizes; (void)n_in; (void)out_size; (void)ws_size;
    const float* x   = (const float*)d_in[0];
    // d_in[1] = attention_mask (causal, implied by kernel)
    const float* Wq  = (const float*)d_in[2];
    const float* Wk  = (const float*)d_in[3];
    const float* Wv  = (const float*)d_in[4];
    const float* Wo  = (const float*)d_in[5];
    const float* bo  = (const float*)d_in[6];
    const float* g1  = (const float*)d_in[7];
    const float* be1 = (const float*)d_in[8];
    const float* g2  = (const float*)d_in[9];
    const float* be2 = (const float*)d_in[10];
    const float* W1  = (const float*)d_in[11];
    const float* b1  = (const float*)d_in[12];
    const float* W2  = (const float*)d_in[13];
    const float* b2  = (const float*)d_in[14];
    float* out = (float*)d_out;

    const size_t HH = (size_t)H_ * H_;
    const size_t HI = (size_t)H_ * I_;
    const size_t MH = (size_t)B_ * S_ * H_;
    const size_t MI = (size_t)B_ * S_ * I_;

    char* ws = (char*)d_ws;
    size_t off = 0;
    auto alloc = [&](size_t bytes) -> void* {
        void* p = ws + off;
        off += (bytes + 255) & ~(size_t)255;
        return p;
    };
    bf16* Wqb = (bf16*)alloc(HH * 2);
    bf16* Wkb = (bf16*)alloc(HH * 2);
    bf16* Wvb = (bf16*)alloc(HH * 2);
    bf16* Wob = (bf16*)alloc(HH * 2);
    bf16* W1b = (bf16*)alloc(HI * 2);
    bf16* W2b = (bf16*)alloc(HI * 2);
    bf16* hb  = (bf16*)alloc(MH * 2);   // LN output (reused for LN1 and LN2)
    bf16* qb  = (bf16*)alloc(MH * 2);
    bf16* kb  = (bf16*)alloc(MH * 2);
    bf16* vb  = (bf16*)alloc(MH * 2);
    bf16* cb  = (bf16*)alloc(MH * 2);   // attention context
    float* x1 = (float*)alloc(MH * 4);  // x + attn_out
    bf16* a1  = (bf16*)alloc(MI * 2);   // gelu(h @ W1 + b1)

    // weights -> bf16
    cvt_kernel<<<2048, 256, 0, stream>>>(Wq, Wqb, (int)HH);
    cvt_kernel<<<2048, 256, 0, stream>>>(Wk, Wkb, (int)HH);
    cvt_kernel<<<2048, 256, 0, stream>>>(Wv, Wvb, (int)HH);
    cvt_kernel<<<2048, 256, 0, stream>>>(Wo, Wob, (int)HH);
    cvt_kernel<<<4096, 256, 0, stream>>>(W1, W1b, (int)HI);
    cvt_kernel<<<4096, 256, 0, stream>>>(W2, W2b, (int)HI);

    const int M = B_ * S_;
    dim3 gH(H_ / 128, M / 128);  // (24, 32)
    dim3 gI(I_ / 128, M / 128);  // (64, 32)

    ln_kernel<<<M, 256, 0, stream>>>(x, g1, be1, hb);
    gemm_bf16_kernel<false, false, false, true><<<gH, 256, 0, stream>>>(
        hb, Wqb, nullptr, nullptr, nullptr, qb, M, H_, H_);
    gemm_bf16_kernel<false, false, false, true><<<gH, 256, 0, stream>>>(
        hb, Wkb, nullptr, nullptr, nullptr, kb, M, H_, H_);
    gemm_bf16_kernel<false, false, false, true><<<gH, 256, 0, stream>>>(
        hb, Wvb, nullptr, nullptr, nullptr, vb, M, H_, H_);

    attn_kernel<<<(B_ * NH_ * (S_ / 16)) / 4, 128, 0, stream>>>(qb, kb, vb, cb);

    gemm_bf16_kernel<true, false, true, false><<<gH, 256, 0, stream>>>(
        cb, Wob, bo, x, x1, nullptr, M, H_, H_);

    ln_kernel<<<M, 256, 0, stream>>>(x1, g2, be2, hb);
    gemm_bf16_kernel<true, true, false, true><<<gI, 256, 0, stream>>>(
        hb, W1b, b1, nullptr, nullptr, a1, M, I_, H_);
    gemm_bf16_kernel<true, false, true, false><<<gH, 256, 0, stream>>>(
        a1, W2b, b2, x1, out, nullptr, M, H_, I_);
}
